// StructuralTransformer_14654428414353
// MI455X (gfx1250) — matrix operand order repaired and checked by reading, not compiled, not run on hardware
//
#include <hip/hip_runtime.h>

// ---------------------------------------------------------------------------
// StructuralTransformer decode loop for MI455X (gfx1250, wave32, WMMA).
//
// Model dims (fixed): D=512, V=1024, L=6, H=8, DH=64, F=2048, S=100, B=8, NQ=8
//
//  * step i runs only on the prefix T=i+1 (mask keys>i == truncate to prefix)
//  * cross-attention has 1 key -> softmax==1 -> per-(layer,batch) constant
//    add vector, precomputed once (ca_Wq/ca_Wk are dead weights)
//  * GEMMs: f16 WMMA (v_wmma_f32_16x16x32_f16), f32 accum; weights converted
//    to f16 once (~46MB, resident in 192MB L2 for all 100 steps)
//  * GEMM register blocking: wave computes 32x32 (2x2 WMMA tiles, 4 accs),
//    block tile 64x128, fragments reused 2x, 4 v_wmma per K-step per wave
// ---------------------------------------------------------------------------

#define D_  512
#define V_  1024
#define L_  6
#define H_  8
#define DH_ 64
#define F_  2048
#define S_  100
#define B_  8
#define NQ_ 8
#define SCALE_ 0.125f   // 1/sqrt(64)

typedef __attribute__((ext_vector_type(16))) _Float16 v16h;
typedef __attribute__((ext_vector_type(2)))  __fp16   v2hf;   // cvt_pkrtz return
typedef __attribute__((ext_vector_type(8)))  float    v8f;

// ---------------------------------------------------------------- f32 -> f16
__global__ void cvt_f16_kernel(const float* __restrict__ src,
                               _Float16* __restrict__ dst, int n) {
  int i = blockIdx.x * 256 + threadIdx.x;
  if (i < n) dst[i] = (_Float16)src[i];
}

// ---------------------------------------------------------------- WMMA GEMM
// C[m,n] = act( sum_k A[m,k] * W[k,n] + bias[n] )
// Row mapping (one kernel handles [B,S,D]-strided activations and the logits
// row-gather):  A row m lives at A + ((m/TR)*SR_A + m%TR)*K + aOff
//               C row m lives at C + ((m/TR)*SR_C + m%TR)*N
// Block tile 64(M) x 128(N); 8 waves as 2(M) x 4(N); each wave owns a 32x32
// output patch = 2x2 v_wmma_f32_16x16x32_f16 tiles; K stepped by 32 via LDS.
__global__ void gemm_wmma_kernel(const float* __restrict__ A,
                                 const _Float16* __restrict__ W,
                                 const float* __restrict__ bias,
                                 float* __restrict__ C,
                                 int M, int N, int K,
                                 int TR, int SR_A, int SR_C,
                                 long long aOff, int relu) {
  __shared__ _Float16 ldsA[64][32];    // [m][k]
  __shared__ _Float16 ldsB[128][32];   // [n][k] (transposed during load)

  const int tid  = threadIdx.x;
  const int m0   = blockIdx.x * 64;
  const int n0   = blockIdx.y * 128;
  const int wave = tid >> 5;
  const int lane = tid & 31;
  const int wm   = wave & 1;           // M half: rows wm*32 .. wm*32+31
  const int wn   = wave >> 1;          // N quarter: cols wn*32 .. wn*32+31
  const int lm   = lane & 15;
  const int hl   = lane >> 4;          // lane half (0/1)

  // ---- per-thread A-tile load mapping (invariant across K loop) ----
  const int ar = tid >> 2;             // row 0..63
  const int ac = (tid & 3) << 3;       // col 0,8,16,24
  const int am = m0 + ar;
  const bool aValid = (am < M);
  long long aBase = 0;
  if (aValid)
    aBase = ((long long)(am / TR) * SR_A + (am % TR)) * K + aOff;

  // ---- per-thread W-tile load mapping (transposed store, packed pairs) ----
  const int kp = tid >> 4;             // k-pair 0..15 -> rows 2kp, 2kp+1
  const int bc = (tid & 15) << 3;      // n chunk 0,8,...,120

  v8f acc00 = {}, acc01 = {}, acc10 = {}, acc11 = {};

  for (int k0 = 0; k0 < K; k0 += 32) {
    __syncthreads();
    {   // A tile: 64x32 f32 -> f16 (v_cvt_pk_rtz), one ds_store_b128
      float4 f0 = make_float4(0.f, 0.f, 0.f, 0.f);
      float4 f1 = make_float4(0.f, 0.f, 0.f, 0.f);
      if (aValid) {
        f0 = *(const float4*)(A + aBase + k0 + ac);
        f1 = *(const float4*)(A + aBase + k0 + ac + 4);
      }
      union { v2hf h2[4]; uint4 q; } pk;
      pk.h2[0] = __builtin_amdgcn_cvt_pkrtz(f0.x, f0.y);
      pk.h2[1] = __builtin_amdgcn_cvt_pkrtz(f0.z, f0.w);
      pk.h2[2] = __builtin_amdgcn_cvt_pkrtz(f1.x, f1.y);
      pk.h2[3] = __builtin_amdgcn_cvt_pkrtz(f1.z, f1.w);
      *(uint4*)&ldsA[ar][ac] = pk.q;
    }
    {   // W tile: rows k0+2kp, k0+2kp+1, 8 n-values each; store packed
        // (W[k][n], W[k+1][n]) u32 pairs transposed -> ldsB[n][k]
      const _Float16* wp0 = W + (long long)(k0 + 2 * kp) * N + n0 + bc;
      const _Float16* wp1 = wp0 + N;
      union { uint4 q; unsigned short s[8]; } r0, r1;
      r0.q = *(const uint4*)wp0;
      r1.q = *(const uint4*)wp1;
      #pragma unroll
      for (int j = 0; j < 8; ++j) {
        unsigned int u = (unsigned int)r0.s[j] | ((unsigned int)r1.s[j] << 16);
        *(unsigned int*)&ldsB[bc + j][2 * kp] = u;
      }
      if (k0 + 32 < K)   // prefetch next W tile (global_prefetch_b8)
        __builtin_prefetch(wp0 + 32LL * N, 0, 3);
    }
    __syncthreads();

    // A fragments: lane half -> K base 0/16;
    // halves e -> K=bA+e (16 consecutive)  => two aligned uint4 LDS loads.
    union { uint4 u4[2]; v16h h; } ua0, ua1;
    const int ao = hl * 8;             // uint offset = bA/2
    {
      const unsigned int* arow = (const unsigned int*)&ldsA[wm * 32 + lm][0];
      ua0.u4[0] = *(const uint4*)(arow + ao);
      ua0.u4[1] = *(const uint4*)(arow + ao + 4);
    }
    {
      const unsigned int* arow = (const unsigned int*)&ldsA[wm * 32 + 16 + lm][0];
      ua1.u4[0] = *(const uint4*)(arow + ao);
      ua1.u4[1] = *(const uint4*)(arow + ao + 4);
    }
    // B fragments (contiguous K, same order as A): lane half -> K base 0/16,
    // 16 consecutive K halves => one 32B LDS load each.
    v16h b0 = *(const v16h*)&ldsB[wn * 32 + lm][hl * 16];
    v16h b1 = *(const v16h*)&ldsB[wn * 32 + 16 + lm][hl * 16];

    acc00 = __builtin_amdgcn_wmma_f32_16x16x32_f16(false, ua0.h, false, b0,
                                                   (short)0, acc00, false, false);
    acc01 = __builtin_amdgcn_wmma_f32_16x16x32_f16(false, ua0.h, false, b1,
                                                   (short)0, acc01, false, false);
    acc10 = __builtin_amdgcn_wmma_f32_16x16x32_f16(false, ua1.h, false, b0,
                                                   (short)0, acc10, false, false);
    acc11 = __builtin_amdgcn_wmma_f32_16x16x32_f16(false, ua1.h, false, b1,
                                                   (short)0, acc11, false, false);
  }

  // Epilogue: C/D layout -> lane holds column n=lm, rows j + 8*hl of each tile
  const int nA = n0 + wn * 32 + lm;
  const int nB = nA + 16;
  const float bvA = bias ? bias[nA] : 0.f;
  const float bvB = bias ? bias[nB] : 0.f;
  #pragma unroll
  for (int ti = 0; ti < 2; ++ti) {
    const v8f* aI0 = ti ? &acc10 : &acc00;
    const v8f* aI1 = ti ? &acc11 : &acc01;
    #pragma unroll
    for (int j = 0; j < 8; ++j) {
      int m = m0 + wm * 32 + ti * 16 + hl * 8 + j;
      if (m < M) {
        long long base = ((long long)(m / TR) * SR_C + (m % TR)) * (long long)N;
        float v0 = (*aI0)[j] + bvA;
        float v1 = (*aI1)[j] + bvB;
        if (relu) { v0 = fmaxf(v0, 0.f); v1 = fmaxf(v1, 0.f); }
        C[base + nA] = v0;
        C[base + nB] = v1;
      }
    }
  }
}

// ----------------------------------------------------- self-attention (f32)
// One block per (b,h). K/V head slices cached in LDS; each wave owns query
// rows qi = wave, wave+8, ... ; softmax via wave shfl reductions.
__global__ void attn_kernel(const float* __restrict__ q,
                            const float* __restrict__ k,
                            const float* __restrict__ v,
                            float* __restrict__ o, int T) {
  __shared__ float Ks[S_][DH_ + 1];
  __shared__ float Vs[S_][DH_ + 1];
  __shared__ float P[8][S_ + 4];
  __shared__ float Qrow[8][DH_ + 2];

  const int b = blockIdx.x >> 3, h = blockIdx.x & 7;
  const int tid = threadIdx.x;

  for (int idx = tid; idx < T * DH_; idx += 256) {
    int t = idx >> 6, d = idx & 63;
    long long g = ((long long)(b * S_ + t)) * D_ + h * DH_ + d;
    Ks[t][d] = k[g];
    Vs[t][d] = v[g];
  }
  __syncthreads();

  const int wave = tid >> 5, lane = tid & 31;
  for (int qi = wave; qi < T; qi += 8) {
    long long gq = ((long long)(b * S_ + qi)) * D_ + h * DH_;
    Qrow[wave][lane]      = q[gq + lane];
    Qrow[wave][lane + 32] = q[gq + lane + 32];
    __builtin_amdgcn_wave_barrier();
    asm volatile("s_wait_dscnt 0" ::: "memory");

    float s[4];
    float mx = -1e30f;
    #pragma unroll
    for (int j = 0; j < 4; ++j) {
      int kk = lane + 32 * j;
      float a = -1e30f;
      if (kk < T) {
        a = 0.f;
        for (int d = 0; d < DH_; ++d) a += Qrow[wave][d] * Ks[kk][d];
        a *= SCALE_;
      }
      s[j] = a;
      mx = fmaxf(mx, a);
    }
    for (int off = 16; off > 0; off >>= 1) mx = fmaxf(mx, __shfl_xor(mx, off, 32));
    float sum = 0.f;
    #pragma unroll
    for (int j = 0; j < 4; ++j) {
      int kk = lane + 32 * j;
      float p = (kk < T) ? __expf(s[j] - mx) : 0.f;
      s[j] = p; sum += p;
    }
    for (int off = 16; off > 0; off >>= 1) sum += __shfl_xor(sum, off, 32);
    float inv = 1.0f / sum;
    #pragma unroll
    for (int j = 0; j < 4; ++j) {
      int kk = lane + 32 * j;
      if (kk < T) P[wave][kk] = s[j] * inv;
    }
    __builtin_amdgcn_wave_barrier();
    asm volatile("s_wait_dscnt 0" ::: "memory");

    float o0 = 0.f, o1 = 0.f;
    for (int kk = 0; kk < T; ++kk) {
      float p = P[wave][kk];
      o0 += p * Vs[kk][lane];
      o1 += p * Vs[kk][lane + 32];
    }
    o[gq + lane]      = o0;
    o[gq + lane + 32] = o1;
  }
}

// ----------------------------------------------- residual add + LayerNorm
// dst[row] = LN(src[row] + add)   ; addPerBatch: add is [B,D] broadcast
__global__ void ln_res_kernel(float* __restrict__ dst,
                              const float* __restrict__ src,
                              const float* __restrict__ add, int addPerBatch,
                              const float* __restrict__ g,
                              const float* __restrict__ beta, int T) {
  __shared__ float red1[8];
  __shared__ float red2[8];
  const int r = blockIdx.x;
  const int b = r / T, t = r % T;
  const long long base = ((long long)(b * S_ + t)) * D_;
  const float* ap = addPerBatch ? (add + (long long)b * D_) : (add + base);
  const int tid = threadIdx.x;

  float x0 = src[base + tid]       + ap[tid];
  float x1 = src[base + tid + 256] + ap[tid + 256];
  float s = x0 + x1;
  for (int off = 16; off > 0; off >>= 1) s += __shfl_xor(s, off, 32);
  if ((tid & 31) == 0) red1[tid >> 5] = s;
  __syncthreads();
  float tot = 0.f;
  #pragma unroll
  for (int i = 0; i < 8; ++i) tot += red1[i];
  float mean = tot * (1.0f / (float)D_);
  float d0 = x0 - mean, d1 = x1 - mean;
  float vs = d0 * d0 + d1 * d1;
  for (int off = 16; off > 0; off >>= 1) vs += __shfl_xor(vs, off, 32);
  if ((tid & 31) == 0) red2[tid >> 5] = vs;
  __syncthreads();
  float vtot = 0.f;
  #pragma unroll
  for (int i = 0; i < 8; ++i) vtot += red2[i];
  float rstd = rsqrtf(vtot * (1.0f / (float)D_) + 1e-5f);
  dst[base + tid]       = d0 * rstd * g[tid]       + beta[tid];
  dst[base + tid + 256] = d1 * rstd * g[tid + 256] + beta[tid + 256];
}

// ----------------------------------- argmax + token store + emb row append
__global__ void argmax_kernel(const float* __restrict__ logits,
                              int* __restrict__ tokens,
                              float* __restrict__ emb,
                              const float* __restrict__ tok_emb, int i) {
  __shared__ float bv[256];
  __shared__ int   bi[256];
  const int b = blockIdx.x, tid = threadIdx.x;
  float best = -1e30f; int bidx = 0x7fffffff;
  for (int j = tid; j < V_; j += 256) {
    float val = logits[b * V_ + j];
    if (val > best) { best = val; bidx = j; }
  }
  bv[tid] = best; bi[tid] = bidx;
  __syncthreads();
  for (int s2 = 128; s2 > 0; s2 >>= 1) {
    if (tid < s2) {
      float ov = bv[tid + s2]; int oi = bi[tid + s2];
      if (ov > bv[tid] || (ov == bv[tid] && oi < bi[tid])) { bv[tid] = ov; bi[tid] = oi; }
    }
    __syncthreads();
  }
  int tok = bi[0];
  if (tid == 0) tokens[b * S_ + i] = tok;
  for (int d = tid; d < D_; d += 256) {
    emb[((long long)(b * S_ + i)) * D_ + d] = tok_emb[(long long)tok * D_ + d];
    if (i + 1 < S_)
      emb[((long long)(b * S_ + i + 1)) * D_ + d] = tok_emb[d];   // placeholder 0
  }
}

__global__ void init_emb_kernel(float* __restrict__ emb,
                                const float* __restrict__ tok_emb) {
  int b = blockIdx.x, tid = threadIdx.x;
  for (int d = tid; d < D_; d += 256)
    emb[((long long)(b * S_)) * D_ + d] = tok_emb[d];             // token 0
}

__global__ void writeout_kernel(const int* __restrict__ tokens,
                                int* __restrict__ out) {
  int idx = blockIdx.x * 256 + threadIdx.x;
  if (idx < B_ * S_ * NQ_) out[idx] = tokens[idx / NQ_];
}

// ---------------------------------------------------------------------------
extern "C" void kernel_launch(void* const* d_in, const int* in_sizes, int n_in,
                              void* d_out, int out_size, void* d_ws, size_t ws_size,
                              hipStream_t stream) {
  (void)in_sizes; (void)n_in; (void)out_size; (void)ws_size;

  // Input order: setup_inputs() insertion order, depth-first.
  const float* ctrl    = (const float*)d_in[0];
  const float* tok_emb = (const float*)d_in[1];
  const float* ctrl_W  = (const float*)d_in[2];
  const float* ctrl_b  = (const float*)d_in[3];
  const float* out_W   = (const float*)d_in[4];
  const float* out_b   = (const float*)d_in[5];
  const float* sa_Wq = (const float*)d_in[6];
  const float* sa_Wk = (const float*)d_in[7];
  const float* sa_Wv = (const float*)d_in[8];
  const float* sa_Wo = (const float*)d_in[9];
  // d_in[10] ca_Wq, d_in[11] ca_Wk: dead (softmax over 1 key == 1)
  const float* ca_Wv = (const float*)d_in[12];
  const float* ca_Wo = (const float*)d_in[13];
  const float* ff_W1 = (const float*)d_in[14];
  const float* ff_W2 = (const float*)d_in[15];
  const float* sa_bq = (const float*)d_in[16];
  const float* sa_bk = (const float*)d_in[17];
  const float* sa_bv = (const float*)d_in[18];
  const float* sa_bo = (const float*)d_in[19];
  const float* ca_bv = (const float*)d_in[22];
  const float* ca_bo = (const float*)d_in[23];
  const float* ff_b1 = (const float*)d_in[24];
  const float* ff_b2 = (const float*)d_in[25];
  const float* ln1_g = (const float*)d_in[26];
  const float* ln1_b = (const float*)d_in[27];
  const float* ln2_g = (const float*)d_in[28];
  const float* ln2_b = (const float*)d_in[29];
  const float* ln3_g = (const float*)d_in[30];
  const float* ln3_b = (const float*)d_in[31];

  // Workspace carving (256B aligned)
  char* ws = (char*)d_ws;
  size_t off = 0;
  auto carve = [&](size_t bytes) -> char* {
    char* p = ws + off;
    off = (off + bytes + 255) & ~(size_t)255;
    return p;
  };
  const size_t WDD = (size_t)L_ * D_ * D_;
  const size_t WDF = (size_t)L_ * D_ * F_;
  _Float16* saWq16  = (_Float16*)carve(WDD * 2);
  _Float16* saWk16  = (_Float16*)carve(WDD * 2);
  _Float16* saWv16  = (_Float16*)carve(WDD * 2);
  _Float16* saWo16  = (_Float16*)carve(WDD * 2);
  _Float16* caWv16  = (_Float16*)carve(WDD * 2);
  _Float16* caWo16  = (_Float16*)carve(WDD * 2);
  _Float16* ffW1_16 = (_Float16*)carve(WDF * 2);
  _Float16* ffW2_16 = (_Float16*)carve(WDF * 2);
  _Float16* ctrlW16 = (_Float16*)carve((size_t)D_ * D_ * 2);
  _Float16* outW16  = (_Float16*)carve((size_t)D_ * V_ * 2);
  float* memBuf = (float*)carve((size_t)B_ * D_ * 4);
  float* memv   = (float*)carve((size_t)B_ * D_ * 4);
  float* caAdd  = (float*)carve((size_t)L_ * B_ * D_ * 4);
  float* emb    = (float*)carve((size_t)B_ * S_ * D_ * 4);
  float* hbuf   = (float*)carve((size_t)B_ * S_ * D_ * 4);
  float* qbuf   = (float*)carve((size_t)B_ * S_ * D_ * 4);
  float* kbuf   = (float*)carve((size_t)B_ * S_ * D_ * 4);
  float* vbuf   = (float*)carve((size_t)B_ * S_ * D_ * 4);
  float* attno  = (float*)carve((size_t)B_ * S_ * D_ * 4);
  float* proj   = (float*)carve((size_t)B_ * S_ * D_ * 4);
  float* ffh    = (float*)carve((size_t)B_ * S_ * F_ * 4);
  float* logits = (float*)carve((size_t)B_ * V_ * 4);
  int*   tokens = (int*)carve((size_t)B_ * S_ * 4);

  auto cvt = [&](const float* s, _Float16* d, size_t n) {
    cvt_f16_kernel<<<dim3((unsigned)((n + 255) / 256)), dim3(256), 0, stream>>>(s, d, (int)n);
  };
  auto gemm = [&](const float* A, const _Float16* W, const float* bias, float* C,
                  int M, int N, int K, int TR, int SRA, int SRC,
                  long long aOff, int relu) {
    dim3 grid((M + 63) / 64, N / 128);
    gemm_wmma_kernel<<<grid, dim3(256), 0, stream>>>(A, W, bias, C, M, N, K,
                                                     TR, SRA, SRC, aOff, relu);
  };

  // 1) one-time weight conversion to f16 (fits in L2; reused by all 100 steps)
  cvt(sa_Wq, saWq16, WDD); cvt(sa_Wk, saWk16, WDD);
  cvt(sa_Wv, saWv16, WDD); cvt(sa_Wo, saWo16, WDD);
  cvt(ca_Wv, caWv16, WDD); cvt(ca_Wo, caWo16, WDD);
  cvt(ff_W1, ffW1_16, WDF); cvt(ff_W2, ffW2_16, WDF);
  cvt(ctrl_W, ctrlW16, (size_t)D_ * D_);
  cvt(out_W,  outW16,  (size_t)D_ * V_);

  // 2) precompute mem and per-(layer,batch) cross-attn add vectors
  gemm(ctrl, ctrlW16, ctrl_b, memBuf, B_, D_, D_, B_, B_, B_, 0, 0);
  for (int l = 0; l < L_; ++l) {
    gemm(memBuf, caWv16 + (size_t)l * D_ * D_, ca_bv + l * D_, memv,
         B_, D_, D_, B_, B_, B_, 0, 0);
    gemm(memv, caWo16 + (size_t)l * D_ * D_, ca_bo + l * D_,
         caAdd + (size_t)l * B_ * D_, B_, D_, D_, B_, B_, B_, 0, 0);
  }

  // 3) seed emb row 0 with token-0 embedding
  init_emb_kernel<<<dim3(B_), dim3(256), 0, stream>>>(emb, tok_emb);

  // 4) decode loop: step i processes prefix T=i+1
  for (int i = 0; i < S_; ++i) {
    const int T = i + 1;
    const int M = B_ * T;
    for (int l = 0; l < L_; ++l) {
      const float* xin = (l == 0) ? emb : hbuf;
      gemm(xin, saWq16 + (size_t)l * D_ * D_, sa_bq + l * D_, qbuf,
           M, D_, D_, T, S_, S_, 0, 0);
      gemm(xin, saWk16 + (size_t)l * D_ * D_, sa_bk + l * D_, kbuf,
           M, D_, D_, T, S_, S_, 0, 0);
      gemm(xin, saWv16 + (size_t)l * D_ * D_, sa_bv + l * D_, vbuf,
           M, D_, D_, T, S_, S_, 0, 0);
      attn_kernel<<<dim3(B_ * H_), dim3(256), 0, stream>>>(qbuf, kbuf, vbuf, attno, T);
      gemm(attno, saWo16 + (size_t)l * D_ * D_, sa_bo + l * D_, proj,
           M, D_, D_, T, S_, S_, 0, 0);
      ln_res_kernel<<<dim3(M), dim3(256), 0, stream>>>(
          hbuf, xin, proj, 0, ln1_g + l * D_, ln1_b + l * D_, T);
      ln_res_kernel<<<dim3(M), dim3(256), 0, stream>>>(
          hbuf, hbuf, caAdd + (size_t)l * B_ * D_, 1,
          ln2_g + l * D_, ln2_b + l * D_, T);
      gemm(hbuf, ffW1_16 + (size_t)l * D_ * F_, ff_b1 + l * F_, ffh,
           M, F_, D_, T, S_, S_, 0, 1 /*relu*/);
      gemm(ffh, ffW2_16 + (size_t)l * F_ * D_, ff_b2 + l * D_, proj,
           M, D_, F_, T, S_, S_, 0, 0);
      ln_res_kernel<<<dim3(M), dim3(256), 0, stream>>>(
          hbuf, hbuf, proj, 0, ln3_g + l * D_, ln3_b + l * D_, T);
    }
    gemm(hbuf, outW16, out_b, logits, B_, V_, D_,
         /*TR=*/1, /*SR_A=*/S_, /*SR_C=*/1, /*aOff=*/(long long)i * D_, 0);
    argmax_kernel<<<dim3(B_), dim3(256), 0, stream>>>(logits, tokens, emb, tok_emb, i);
  }

  // 5) tokens -> d_out repeated NQ times (int32 output)
  writeout_kernel<<<dim3((B_ * S_ * NQ_ + 255) / 256), dim3(256), 0, stream>>>(
      tokens, (int*)d_out);
}
